// KNNConnector_2491081031888
// MI455X (gfx1250) — compile-verified, matrix operand order checked by reading, not run on hardware
//
#include <hip/hip_runtime.h>
#include <stdint.h>

// ---------------------------------------------------------------------------
// KNN connector for MI455X (gfx1250).
//   d(i,j) ranking uses  ||p_j||^2 - 2 p_i . p_j   (row norm is rank-invariant)
//   Gram tiles via v_wmma_f32_16x16x32_bf16 with split-bf16 error compensation:
//   p = hi + lo (both bf16);  p_i . p_j ~= hi.hi + hi.lo + lo.hi  (3 WMMAs)
// ---------------------------------------------------------------------------

#define N_NODES     16384
#define DIMS        256
#define KNN         32
#define ROWS_PER_WG 64
#define THREADS     256          // 8 wave32s
#define STAGE_COLS  128          // 8 waves x 16-col tile
#define N_STAGES    (N_NODES / STAGE_COLS)
#define SUBS        4            // selection subranges per row
#define SUB_COLS    (STAGE_COLS / SUBS)
#define KSTEPS      (DIMS / 32)

typedef __attribute__((ext_vector_type(16))) __bf16 v16bf;
typedef __attribute__((ext_vector_type(8)))  float  v8f;

union Frag16 {            // 16 bf16 = 32 bytes = two 128-bit chunks
    v16bf v;
    uint4 q[2];
};

__device__ __forceinline__ uint16_t f32_to_bf16_rne(float x) {
    uint32_t b = __float_as_uint(x);
    uint32_t r = b + 0x7FFFu + ((b >> 16) & 1u);   // round-to-nearest-even
    return (uint16_t)(r >> 16);
}

// ---------------------------------------------------------------------------
// Kernel 1: split p into bf16 hi/lo and compute per-row squared norms.
// One block per row, one dim per thread (DIMS == THREADS).
// ---------------------------------------------------------------------------
__global__ __launch_bounds__(THREADS)
void knn_prep_kernel(const float* __restrict__ p,
                     uint16_t* __restrict__ pHi,
                     uint16_t* __restrict__ pLo,
                     float* __restrict__ sq) {
    __shared__ float red[THREADS];
    const int row = blockIdx.x;
    const int t   = threadIdx.x;

    const float x = p[row * DIMS + t];

    uint32_t b      = __float_as_uint(x);
    uint32_t r      = b + 0x7FFFu + ((b >> 16) & 1u);
    uint32_t hibits = r & 0xFFFF0000u;
    float    fhi    = __uint_as_float(hibits);
    float    lo     = x - fhi;

    pHi[row * DIMS + t] = (uint16_t)(hibits >> 16);
    pLo[row * DIMS + t] = f32_to_bf16_rne(lo);

    red[t] = x * x;
    __syncthreads();
    for (int s = THREADS / 2; s > 0; s >>= 1) {
        if (t < s) red[t] += red[t + s];
        __syncthreads();
    }
    if (t == 0) sq[row] = red[0];
}

// ---------------------------------------------------------------------------
// Kernel 2: per 64-row block, stream all 16384 columns in 128-col stages.
//   compute phase : 8 waves x (4 row tiles x 8 k-steps x 3 WMMAs)
//   select phase  : 256 threads = 64 rows x 4 subranges, threshold top-32
// ---------------------------------------------------------------------------
__global__ __launch_bounds__(THREADS)
void knn_main_kernel(const uint16_t* __restrict__ pHi,
                     const uint16_t* __restrict__ pLo,
                     const float* __restrict__ sq,
                     const unsigned char* __restrict__ active,
                     int* __restrict__ out) {
    __shared__ uint16_t sAhi [ROWS_PER_WG * DIMS];        // 32 KB
    __shared__ uint16_t sAlo [ROWS_PER_WG * DIMS];        // 32 KB
    __shared__ float    sTile[ROWS_PER_WG * STAGE_COLS];  // 32 KB
    __shared__ float    sDist[THREADS * KNN];             // 32 KB
    __shared__ int      sIdx [THREADS * KNN];             // 32 KB

    const int tid      = threadIdx.x;
    const int lane     = tid & 31;
    const int wave     = tid >> 5;
    const int laneM    = lane & 15;            // M / N position within tile
    const int laneHalf = (lane >> 4) << 3;     // 0 or 8 (K/M half select)
    const int wgRow    = blockIdx.x * ROWS_PER_WG;

    // ---- stage the 64-row A block (bf16 hi/lo) into LDS, 128-bit copies ----
    {
        const uint4* gH = (const uint4*)(pHi + (size_t)wgRow * DIMS);
        const uint4* gL = (const uint4*)(pLo + (size_t)wgRow * DIMS);
        uint4* lH = (uint4*)sAhi;
        uint4* lL = (uint4*)sAlo;
        #pragma unroll
        for (int i = 0; i < (ROWS_PER_WG * DIMS / 8) / THREADS; ++i) {
            lH[i * THREADS + tid] = gH[i * THREADS + tid];
            lL[i * THREADS + tid] = gL[i * THREADS + tid];
        }
    }

    // ---- init private top-k lists (in LDS) ----
    float curWorst  = 3.4e38f;
    int   worstSlot = 0;
    for (int s = 0; s < KNN; ++s) {
        sDist[tid * KNN + s] = 3.4e38f;
        sIdx [tid * KNN + s] = N_NODES - 1;
    }
    __syncthreads();

    for (int stage = 0; stage < N_STAGES; ++stage) {
        // ============== compute phase: one 16-col tile per wave ==============
        const int colBase = stage * STAGE_COLS + wave * 16;
        const int bcol    = colBase + laneM;
        const uint16_t* bRowH = pHi + (size_t)bcol * DIMS;
        const uint16_t* bRowL = pLo + (size_t)bcol * DIMS;

        v8f acc[4] = {v8f{}, v8f{}, v8f{}, v8f{}};

        #pragma unroll
        for (int ks = 0; ks < KSTEPS; ++ks) {
            const int koff = ks * 32 + laneHalf;
            Frag16 bh, bl;
            bh.q[0] = *(const uint4*)(bRowH + koff);
            bh.q[1] = *(const uint4*)(bRowH + koff + 16);
            bl.q[0] = *(const uint4*)(bRowL + koff);
            bl.q[1] = *(const uint4*)(bRowL + koff + 16);

            #pragma unroll
            for (int rt = 0; rt < 4; ++rt) {
                const int aoff = (rt * 16 + laneM) * DIMS + koff;
                Frag16 ah, al;
                ah.q[0] = *(const uint4*)(sAhi + aoff);
                ah.q[1] = *(const uint4*)(sAhi + aoff + 16);
                al.q[0] = *(const uint4*)(sAlo + aoff);
                al.q[1] = *(const uint4*)(sAlo + aoff + 16);

                acc[rt] = __builtin_amdgcn_wmma_f32_16x16x32_bf16(
                    false, ah.v, false, bh.v, (short)0, acc[rt], false, false);
                acc[rt] = __builtin_amdgcn_wmma_f32_16x16x32_bf16(
                    false, ah.v, false, bl.v, (short)0, acc[rt], false, false);
                acc[rt] = __builtin_amdgcn_wmma_f32_16x16x32_bf16(
                    false, al.v, false, bh.v, (short)0, acc[rt], false, false);
            }
        }

        // distances = ||col||^2 - 2 * dot  -> LDS staging tile
        const float sqc = sq[bcol];
        #pragma unroll
        for (int rt = 0; rt < 4; ++rt) {
            #pragma unroll
            for (int r = 0; r < 8; ++r) {
                const int rowLoc = rt * 16 + laneHalf + r;
                sTile[rowLoc * STAGE_COLS + wave * 16 + laneM] =
                    fmaf(-2.0f, acc[rt][r], sqc);
            }
        }

        // prefetch next stage's B rows into cache (global_prefetch_b8)
        if (stage + 1 < N_STAGES) {
            __builtin_prefetch(pHi + (size_t)(bcol + STAGE_COLS) * DIMS, 0, 0);
            __builtin_prefetch(pLo + (size_t)(bcol + STAGE_COLS) * DIMS, 0, 0);
        }

        __syncthreads();

        // ============== select phase: 64 rows x 4 subranges ==============
        {
            const int rsel  = tid & (ROWS_PER_WG - 1);
            const int sub   = tid >> 6;                  // 0..3
            const int lbase = tid * KNN;
            #pragma unroll 4
            for (int j = 0; j < SUB_COLS; ++j) {
                const int   cl = sub * SUB_COLS + j;
                const float d  = sTile[rsel * STAGE_COLS + cl];
                if (d < curWorst) {
                    sDist[lbase + worstSlot] = d;
                    sIdx [lbase + worstSlot] = stage * STAGE_COLS + cl;
                    float w  = -3.4e38f;
                    int   ws = 0;
                    for (int s = 0; s < KNN; ++s) {
                        const float v = sDist[lbase + s];
                        if (v > w) { w = v; ws = s; }
                    }
                    curWorst  = w;
                    worstSlot = ws;
                }
            }
        }
        __syncthreads();
    }

    // ---- merge 4 partial lists per row -> final top-32, emit edges ----
    if (tid < ROWS_PER_WG) {
        const int  row       = tid;
        const int  rowGlobal = wgRow + row;
        const bool act       = active[rowGlobal] != 0;
        const int  sentinel  = N_NODES - 1;
        for (int j = 0; j < KNN; ++j) {
            float best = 3.4e38f;
            int   bt = row, bs = 0;
            #pragma unroll
            for (int t4 = 0; t4 < SUBS; ++t4) {
                const int tt = t4 * ROWS_PER_WG + row;
                for (int s = 0; s < KNN; ++s) {
                    const float v = sDist[tt * KNN + s];
                    if (v < best) { best = v; bt = tt; bs = s; }
                }
            }
            const int bi = sIdx[bt * KNN + bs];
            sDist[bt * KNN + bs] = 3.4e38f;              // consume
            out[rowGlobal * KNN + j]                     = act ? bi        : sentinel;
            out[N_NODES * KNN + rowGlobal * KNN + j]     = act ? rowGlobal : sentinel;
        }
    }
}

// ---------------------------------------------------------------------------
extern "C" void kernel_launch(void* const* d_in, const int* in_sizes, int n_in,
                              void* d_out, int out_size, void* d_ws, size_t ws_size,
                              hipStream_t stream) {
    const float*         p      = (const float*)d_in[0];
    const unsigned char* active = (const unsigned char*)d_in[1];
    // d_in[2] is k == 32, baked into KNN.

    uint16_t* pHi = (uint16_t*)d_ws;                                 // 8 MB
    uint16_t* pLo = pHi + (size_t)N_NODES * DIMS;                    // 8 MB
    float*    sq  = (float*)(pLo + (size_t)N_NODES * DIMS);          // 64 KB
    int*      out = (int*)d_out;                                     // senders | receivers

    knn_prep_kernel<<<N_NODES, THREADS, 0, stream>>>(p, pHi, pLo, sq);
    knn_main_kernel<<<N_NODES / ROWS_PER_WG, THREADS, 0, stream>>>(pHi, pLo, sq, active, out);
}
